// Token_MS_Attention_5196910428186
// MI455X (gfx1250) — matrix operand order repaired and checked by reading, not compiled, not run on hardware
//
#include <hip/hip_runtime.h>

// ---------------------------------------------------------------------------
// CDNA5 (gfx1250) fused spiking token-attention block, round 3.
//  * xT / yT pre-transposed bf16 [t,b,n,c] so all LDS staging is 16B block
//    copies (async global->LDS via GLOBAL_LOAD_ASYNC_TO_LDS_B128, ASYNCcnt).
//  * k-loop outermost, t innermost: 4 t-accumulators in VGPRs -> 20 WMMAs per
//    A-fragment fetch in the attn kernel; LIF scan is a register epilogue.
// ---------------------------------------------------------------------------

typedef __attribute__((ext_vector_type(16))) __bf16 v16bf;
typedef __attribute__((ext_vector_type(8)))  __bf16 v8bf;
typedef __attribute__((ext_vector_type(8)))  float  v8f;
typedef __attribute__((ext_vector_type(4)))  int    v4i;
typedef __attribute__((address_space(1))) v4i gv4i;   // global int4
typedef __attribute__((address_space(3))) v4i sv4i;   // LDS int4

#define Cdim   512
#define Npix   256
#define Bdim   32
#define Tdim   4
#define MAT_ELEMS (Cdim * Cdim)     // elements per swizzled weight matrix
#define XS_STRIDE 48                // bf16 per LDS row (96B: keeps 32B frags aligned)

#if defined(__has_builtin)
#  if __has_builtin(__builtin_amdgcn_global_load_async_to_lds_b128)
#    define USE_ASYNC_LDS 1
#  endif
#endif
#ifndef USE_ASYNC_LDS
#  define USE_ASYNC_LDS 0
#endif

static __device__ __forceinline__ v8f wmma_bf16(v16bf a, v16bf b, v8f c) {
  return __builtin_amdgcn_wmma_f32_16x16x32_bf16(false, a, false, b,
                                                 (short)0, c, false, false);
}

// 16-byte global -> LDS staging copy (async if the toolchain exposes it).
static __device__ __forceinline__ void stage16(__bf16* dst, const __bf16* src) {
#if USE_ASYNC_LDS
  __builtin_amdgcn_global_load_async_to_lds_b128((gv4i*)src, (sv4i*)dst, 0, 0);
#else
  *(v8bf*)dst = *(const v8bf*)src;
#endif
}

static __device__ __forceinline__ void stage_fence() {
#if USE_ASYNC_LDS
#  if __has_builtin(__builtin_amdgcn_s_wait_asynccnt)
  __builtin_amdgcn_s_wait_asynccnt(0);
#  else
  asm volatile("s_wait_asynccnt 0" ::: "memory");
#  endif
#endif
}

// ---------------------------------------------------------------------------
// Weight swizzle: [d,c] f32 -> bf16 in A-fragment lane order
// (contiguous k = 16*(lane/16) + e, matching the B reads). Matrices: 0=qW 1=kW 2..4=pW taps 5=projW.
// ---------------------------------------------------------------------------
__global__ void prep_weights_k(const float* __restrict__ qW,
                               const float* __restrict__ kW,
                               const float* __restrict__ pW,
                               const float* __restrict__ projW,
                               __bf16* __restrict__ dst) {
  const int mat = blockIdx.y;
  const int idx = blockIdx.x * blockDim.x + threadIdx.x;   // < 262144
  const int f      = idx >> 9;
  const int within = idx & 511;
  const int lane   = within >> 4;
  const int e      = within & 15;
  const int mT = f >> 4;
  const int kT = f & 15;
  const int d  = mT * 16 + (lane & 15);
  const int c  = kT * 32 + ((lane >= 16) ? 16 : 0) + e;
  float v;
  if      (mat == 0) v = qW[d * Cdim + c];
  else if (mat == 1) v = kW[d * Cdim + c];
  else if (mat <= 4) v = pW[(d * Cdim + c) * 3 + (mat - 2)];
  else               v = projW[d * Cdim + c];
  dst[mat * MAT_ELEMS + idx] = (__bf16)v;
}

// BN fold: scale = g/sqrt(v+eps); shift = beta - mean*scale (+bias*scale, proj)
__global__ void prep_bn_k(const float* __restrict__ qbn, const float* __restrict__ kbn,
                          const float* __restrict__ pbn, const float* __restrict__ projbn,
                          const float* __restrict__ projb, float* __restrict__ out) {
  const int i  = blockIdx.x * blockDim.x + threadIdx.x;    // < 4*C
  if (i >= 4 * Cdim) return;
  const int br = i >> 9;
  const int c  = i & 511;
  const float* p = (br == 0) ? qbn : (br == 1) ? kbn : (br == 2) ? pbn : projbn;
  const float g = p[c], b = p[Cdim + c], m = p[2 * Cdim + c], v = p[3 * Cdim + c];
  const float sc = g * rsqrtf(v + 1e-5f);
  float sh = b - m * sc;
  if (br == 3) sh += projb[c] * sc;
  out[br * 2 * Cdim + c]        = sc;
  out[br * 2 * Cdim + Cdim + c] = sh;
}

// ---------------------------------------------------------------------------
// x [t,b,c,n] f32 -> xT [t,b,n,c] bf16 (LDS-tiled 32x32 transpose, coalesced).
// grid = (N/32, C/32, T*B), block = 256.
// ---------------------------------------------------------------------------
__global__ __launch_bounds__(256)
void xpose_k(const float* __restrict__ x, __bf16* __restrict__ xT) {
  __shared__ __bf16 tile[32 * 34];
  const int n0 = blockIdx.x * 32;
  const int c0 = blockIdx.y * 32;
  const int tb = blockIdx.z;
  const int tl = threadIdx.x & 31;
  const int th = threadIdx.x >> 5;
#pragma unroll
  for (int p = 0; p < 4; ++p) {
    const int cl = th + p * 8;
    tile[tl * 34 + cl] =
        (__bf16)x[((size_t)tb * Cdim + c0 + cl) * Npix + n0 + tl];
  }
  __syncthreads();
#pragma unroll
  for (int p = 0; p < 4; ++p) {
    const int nl = th + p * 8;
    xT[((size_t)tb * Npix + n0 + nl) * Cdim + c0 + tl] = tile[nl * 34 + tl];
  }
}

// ---------------------------------------------------------------------------
// Fused q/k/p GEMMs + BN + LIF + head-sum + attn-LIF + y = attn*k  (bf16 yT)
// grid = (N/32, heads, B), block = 256 (8 waves).
// k-loop outermost; per k-step: 5 A-fragments, 3 B-fragments x 4t, 20 WMMAs.
// ---------------------------------------------------------------------------
__global__ __launch_bounds__(256)
void attn_qkp_kernel(const __bf16* __restrict__ xT, const __bf16* __restrict__ aw,
                     const float* __restrict__ bn, __bf16* __restrict__ yT) {
  const int ntile = blockIdx.x;           // 0..7
  const int head  = blockIdx.y;           // 0..7
  const int b     = blockIdx.z;           // 0..31
  const int tid   = threadIdx.x;
  const int lane  = tid & 31;
  const int wave  = tid >> 5;
  const int mTile = wave >> 1;            // 0..3 -> 16 head channels
  const int nTsub = wave & 1;             // 0..1 -> 16 columns
  const int hi    = lane >> 4;
  const int nIdx  = nTsub * 16 + (lane & 15);
  const int n0    = ntile * 32;

  // [t][haloCol(34)][c-block], rows 96B so 32B fragment reads stay aligned
  __shared__ __attribute__((aligned(32))) __bf16 xs[Tdim * 34 * XS_STRIDE];
  __shared__ float red[32], avs[32], asp[32];

  if (tid < 32) avs[tid] = 0.0f;          // attn membrane per n

  const __bf16* awq  = aw;
  const __bf16* awk  = aw + 1 * MAT_ELEMS;
  const __bf16* awp0 = aw + 2 * MAT_ELEMS;
  const __bf16* awp1 = aw + 3 * MAT_ELEMS;
  const __bf16* awp2 = aw + 4 * MAT_ELEMS;

  v8f qa[Tdim] = {}, ka[Tdim] = {}, pa[Tdim] = {};
  const int mG = head * 4 + mTile;

  for (int kt = 0; kt < 16; ++kt) {       // K = 512 in steps of 32
    const int cb = kt * 32;
    __syncthreads();                      // xs consumed by previous step

    // main columns n0..n0+31: 4t*32col*4chunks = 512 16B copies, no div/mod
#pragma unroll
    for (int it = 0; it < 2; ++it) {
      const int idx = tid + it * 256;
      const int tt  = idx >> 7;
      const int col = (idx >> 2) & 31;
      const int q   = idx & 3;
      stage16(xs + ((tt * 34 + col + 1) * XS_STRIDE) + q * 8,
              xT + ((size_t)((tt * Bdim + b) * Npix + n0 + col)) * Cdim + cb + q * 8);
    }
    // halo columns n0-1 and n0+32 (zero pad at image edges)
    if (tid < 32) {
      const int q  = tid & 3;
      const int tt = (tid >> 2) & 3;
      const int ch = tid >> 4;            // 0 -> left halo, 1 -> right halo
      const int col = ch ? 33 : 0;
      const int n   = ch ? (n0 + 32) : (n0 - 1);
      __bf16* dst = xs + (tt * 34 + col) * XS_STRIDE + q * 8;
      if (n < 0 || n >= Npix) {
        v8bf z = {};
        *(v8bf*)dst = z;
      } else {
        stage16(dst, xT + ((size_t)((tt * Bdim + b) * Npix + n)) * Cdim + cb + q * 8);
      }
    }
    stage_fence();
    __syncthreads();

    // A fragments: one contiguous 32B load per lane per matrix (L2-hot)
    const int fo = (mG * 16 + kt) * 512 + lane * 16;
    const v16bf aq  = *(const v16bf*)(awq  + fo);
    const v16bf ak  = *(const v16bf*)(awk  + fo);
    const v16bf ap0 = *(const v16bf*)(awp0 + fo);
    const v16bf ap1 = *(const v16bf*)(awp1 + fo);
    const v16bf ap2 = *(const v16bf*)(awp2 + fo);

#pragma unroll
    for (int tt = 0; tt < Tdim; ++tt) {
      const __bf16* base = xs + tt * 34 * XS_STRIDE;
      const int r0 = 1 + nIdx;
      const v16bf bm = *(const v16bf*)(base + (r0 - 1) * XS_STRIDE + hi * 16);
      const v16bf bc = *(const v16bf*)(base + (r0    ) * XS_STRIDE + hi * 16);
      const v16bf bp = *(const v16bf*)(base + (r0 + 1) * XS_STRIDE + hi * 16);
      qa[tt] = wmma_bf16(aq,  bc, qa[tt]);
      ka[tt] = wmma_bf16(ak,  bc, ka[tt]);
      pa[tt] = wmma_bf16(ap0, bm, pa[tt]);
      pa[tt] = wmma_bf16(ap1, bc, pa[tt]);
      pa[tt] = wmma_bf16(ap2, bp, pa[tt]);
    }
  }

  // ---- epilogue: BN + LIF temporal scan + head-sum + attn LIF + y ----
  float qsc[8], qsh[8], ksc[8], ksh[8], psc[8], psh[8];
  const int dBase = head * 64 + mTile * 16;
#pragma unroll
  for (int v = 0; v < 8; ++v) {
    const int d = dBase + v + hi * 8;
    qsc[v] = bn[d];              qsh[v] = bn[Cdim + d];
    ksc[v] = bn[2 * Cdim + d];   ksh[v] = bn[3 * Cdim + d];
    psc[v] = bn[4 * Cdim + d];   psh[v] = bn[5 * Cdim + d];
  }

  v8f vq = {}, vk = {}, vp = {};
#pragma unroll
  for (int tt = 0; tt < Tdim; ++tt) {
    float ks_frag[8];
    float qp_partial = 0.0f;
#pragma unroll
    for (int v = 0; v < 8; ++v) {
      float pre = qa[tt][v] * qsc[v] + qsh[v];
      vq[v] += (pre - vq[v]) * 0.5f;                 // 1/TAU
      const float sq = (vq[v] >= 1.0f) ? 1.0f : 0.0f;
      vq[v] *= (1.0f - sq);

      pre = ka[tt][v] * ksc[v] + ksh[v];
      vk[v] += (pre - vk[v]) * 0.5f;
      const float sk = (vk[v] >= 1.0f) ? 1.0f : 0.0f;
      vk[v] *= (1.0f - sk);
      ks_frag[v] = sk;

      pre = pa[tt][v] * psc[v] + psh[v];
      vp[v] += (pre - vp[v]) * 0.5f;
      const float sp = (vp[v] >= 1.0f) ? 1.0f : 0.0f;
      vp[v] *= (1.0f - sp);

      qp_partial += sq + sp;
    }

    __syncthreads();
    if (tid < 32) red[tid] = 0.0f;
    __syncthreads();
    atomicAdd(&red[nIdx], qp_partial);    // sum over the head's 64 channels
    __syncthreads();
    if (tid < 32) {                        // attn LIF per n, vth = 0.5
      const float pre = red[tid];
      float av = avs[tid];
      av += (pre - av) * 0.5f;
      const float s = (av >= 0.5f) ? 1.0f : 0.0f;
      avs[tid] = av * (1.0f - s);
      asp[tid] = s;
    }
    __syncthreads();
    const float a_s = asp[nIdx];

    v8bf yv;
#pragma unroll
    for (int v = 0; v < 8; ++v) yv[v] = (__bf16)(a_s * ks_frag[v]);
    // lane owns 8 consecutive channels -> one 16B store, [t,b,n,c] layout
    *(v8bf*)(yT + ((size_t)((tt * Bdim + b) * Npix + n0 + nIdx)) * Cdim +
             dBase + hi * 8) = yv;
  }
}

// ---------------------------------------------------------------------------
// proj GEMM over yT (K=512) + folded bias/BN + LIF -> f32 spikes to d_out
// grid = (N/32, C/64, B), block = 256; k outer, 4 t-accumulators.
// ---------------------------------------------------------------------------
__global__ __launch_bounds__(256)
void proj_kernel(const __bf16* __restrict__ yT, const __bf16* __restrict__ awproj,
                 const float* __restrict__ bn, float* __restrict__ out) {
  const int ntile = blockIdx.x;
  const int dTile = blockIdx.y;
  const int b     = blockIdx.z;
  const int tid   = threadIdx.x;
  const int lane  = tid & 31;
  const int wave  = tid >> 5;
  const int mTile = wave >> 1;
  const int nTsub = wave & 1;
  const int hi    = lane >> 4;
  const int nIdx  = nTsub * 16 + (lane & 15);
  const int n0    = ntile * 32;

  __shared__ __attribute__((aligned(32))) __bf16 ys[Tdim * 32 * XS_STRIDE];

  v8f acc[Tdim] = {};
  const int mG = dTile * 4 + mTile;

  for (int kt = 0; kt < 16; ++kt) {
    const int cb = kt * 32;
    __syncthreads();
#pragma unroll
    for (int it = 0; it < 2; ++it) {
      const int idx = tid + it * 256;
      const int tt  = idx >> 7;
      const int col = (idx >> 2) & 31;
      const int q   = idx & 3;
      stage16(ys + ((tt * 32 + col) * XS_STRIDE) + q * 8,
              yT + ((size_t)((tt * Bdim + b) * Npix + n0 + col)) * Cdim + cb + q * 8);
    }
    stage_fence();
    __syncthreads();

    const int fo = (mG * 16 + kt) * 512 + lane * 16;
    const v16bf a = *(const v16bf*)(awproj + fo);
#pragma unroll
    for (int tt = 0; tt < Tdim; ++tt) {
      const v16bf bb =
          *(const v16bf*)(ys + (tt * 32 + nIdx) * XS_STRIDE + hi * 16);
      acc[tt] = wmma_bf16(a, bb, acc[tt]);
    }
  }

  float sc[8], sh[8];
  const int dBase = dTile * 64 + mTile * 16;
#pragma unroll
  for (int v = 0; v < 8; ++v) {
    const int d = dBase + v + hi * 8;
    sc[v] = bn[6 * Cdim + d];
    sh[v] = bn[7 * Cdim + d];
  }

  v8f vm = {};
  const int n = n0 + nIdx;
#pragma unroll
  for (int tt = 0; tt < Tdim; ++tt) {
#pragma unroll
    for (int v = 0; v < 8; ++v) {
      const float pre = acc[tt][v] * sc[v] + sh[v];
      vm[v] += (pre - vm[v]) * 0.5f;
      const float s = (vm[v] >= 1.0f) ? 1.0f : 0.0f;
      vm[v] *= (1.0f - s);
      out[(((size_t)(tt * Bdim + b) * Cdim) + dBase + v + hi * 8) * Npix + n] = s;
    }
  }
}

// ---------------------------------------------------------------------------
extern "C" void kernel_launch(void* const* d_in, const int* in_sizes, int n_in,
                              void* d_out, int out_size, void* d_ws, size_t ws_size,
                              hipStream_t stream) {
  const float* x       = (const float*)d_in[0];
  const float* qW      = (const float*)d_in[1];
  const float* q_bn    = (const float*)d_in[2];
  const float* kW      = (const float*)d_in[3];
  const float* k_bn    = (const float*)d_in[4];
  const float* pW      = (const float*)d_in[5];
  const float* p_bn    = (const float*)d_in[6];
  const float* projW   = (const float*)d_in[7];
  const float* proj_b  = (const float*)d_in[8];
  const float* proj_bn = (const float*)d_in[9];
  float* out = (float*)d_out;

  // workspace layout
  const size_t AW_BYTES = (size_t)6 * MAT_ELEMS * sizeof(__bf16);          // 3 MB
  const size_t BN_BYTES = (size_t)8 * Cdim * sizeof(float);                // 16 KB
  const size_t XT_BYTES = (size_t)Tdim * Bdim * Npix * Cdim * sizeof(__bf16); // 33.5 MB
  __bf16* aw    = (__bf16*)d_ws;
  float*  bnbuf = (float*)((char*)d_ws + AW_BYTES);
  __bf16* xTbuf = (__bf16*)((char*)d_ws + AW_BYTES + BN_BYTES);
  __bf16* yTbuf = (__bf16*)((char*)d_ws + AW_BYTES + BN_BYTES + XT_BYTES);

  prep_weights_k<<<dim3(MAT_ELEMS / 256, 6), 256, 0, stream>>>(qW, kW, pW, projW, aw);
  prep_bn_k<<<dim3((4 * Cdim) / 256), 256, 0, stream>>>(q_bn, k_bn, p_bn,
                                                        proj_bn, proj_b, bnbuf);
  xpose_k<<<dim3(Npix / 32, Cdim / 32, Tdim * Bdim), 256, 0, stream>>>(x, xTbuf);
  attn_qkp_kernel<<<dim3(Npix / 32, 8, Bdim), 256, 0, stream>>>(xTbuf, aw, bnbuf, yTbuf);
  proj_kernel<<<dim3(Npix / 32, Cdim / 64, Bdim), 256, 0, stream>>>(
      yTbuf, aw + 5 * MAT_ELEMS, bnbuf, out);
}
